// RNN_27032524161350
// MI455X (gfx1250) — matrix operand order repaired and checked by reading, not compiled, not run on hardware
//
#include <hip/hip_runtime.h>

// ---------------------------------------------------------------------------
// 2-layer tanh RNN for MI455X (gfx1250, wave32, WMMA, async-LDS, TDM).
//
//   1. weights f32->bf16, bias combine
//   2. gather embeddings -> bf16 x [B*S, EMB]
//   3. GEMM (WMMA bf16, async double-buffered LDS): xp = x @ W_ih0^T + b0
//   4. persistent recurrence L0: 16 WGs, W_hh slice TDM'd into LDS, grid-sync
//   5. GEMM: xp = h0 @ W_ih1^T + b1
//   6. persistent recurrence L1
//   7. GEMM: out = h1 @ W_out^T + b_out
// ---------------------------------------------------------------------------

#define B_    32
#define S_    512
#define EMB_  512
#define HID_  1024
#define LAB_  64
#define NWG_REC 16            // HID_/64 hidden-column slices

typedef __attribute__((ext_vector_type(16))) __bf16 v16bf;
typedef __attribute__((ext_vector_type(8)))  __bf16 v8bf;
typedef __attribute__((ext_vector_type(8)))  float  v8f;
typedef __attribute__((ext_vector_type(4)))  unsigned u32x4;
typedef __attribute__((ext_vector_type(8)))  unsigned u32x8;

// ---- CDNA5 wait helpers ---------------------------------------------------
__device__ __forceinline__ void wait_async0() {
#if __has_builtin(__builtin_amdgcn_s_wait_asynccnt)
  __builtin_amdgcn_s_wait_asynccnt(0);
#else
  asm volatile("s_wait_asynccnt 0x0" ::: "memory");
#endif
}
__device__ __forceinline__ void wait_tensor0() {
#if __has_builtin(__builtin_amdgcn_s_wait_tensorcnt)
  __builtin_amdgcn_s_wait_tensorcnt(0);
#else
  asm volatile("s_wait_tensorcnt 0x0" ::: "memory");
#endif
}
// low 32 bits of a generic pointer to LDS == wave-relative LDS byte address
__device__ __forceinline__ unsigned lds_off(const void* p) {
  return (unsigned)(uintptr_t)p;
}
// async 16B copy global -> LDS (GVS mode: SGPR base + 32-bit VGPR offset)
__device__ __forceinline__ void async_ld_b128(unsigned lds, unsigned goff,
                                              const void* base) {
  asm volatile("global_load_async_to_lds_b128 %0, %1, %2"
               :: "v"(lds), "v"(goff),
                  "s"((unsigned long long)(uintptr_t)base)
               : "memory");
}

// ---- WMMA fragment loaders (contiguous k order, same for A and B) ----------
__device__ __forceinline__ v16bf make_v16(v8bf lo, v8bf hi) {
  return __builtin_shufflevector(lo, hi, 0,1,2,3,4,5,6,7,8,9,10,11,12,13,14,15);
}
// A (16x32, MxK): lane halves split K 0..15/16..31.
__device__ __forceinline__ v16bf load_frag_A(const __bf16* base, int ld, int lane) {
  const int half = lane >> 4;
  const int r    = lane & 15;
  const __bf16* p = base + r * ld + half * 16;
  return make_v16(*(const v8bf*)(p), *(const v8bf*)(p + 8));
}
// B (32x16, KxN) from W stored [N,K] row-major: lane halves split K 0..15/16..31.
__device__ __forceinline__ v16bf load_frag_B(const __bf16* base, int ld, int lane) {
  const int half = lane >> 4;
  const int n    = lane & 15;
  const __bf16* p = base + n * ld + half * 16;
  return make_v16(*(const v8bf*)(p), *(const v8bf*)(p + 8));
}

// ---------------------------------------------------------------------------
// GEMM: C[M,N] = A[M,K](bf16) @ W[N,K](bf16)^T + bias[N]   (f32 out)
// Block 128x64, 8 waves (4x2), wave tile 32x32. Async double-buffered LDS.
// ---------------------------------------------------------------------------
#define BM 128
#define BN 64
#define BK 32
#define LDT 40   // padded LDS stride: 80B rows, 16B-aligned, conflict-free

__global__ __launch_bounds__(256)
void gemm_bf16_wmma(const __bf16* __restrict__ A, const __bf16* __restrict__ W,
                    const float* __restrict__ bias, float* __restrict__ C,
                    int M, int N, int K) {
  __shared__ __bf16 lsA[2][BM * LDT];
  __shared__ __bf16 lsW[2][BN * LDT];
  const int tid  = threadIdx.x;
  const int lane = tid & 31;
  const int w    = tid >> 5;
  const int wm   = w >> 1;      // 0..3
  const int wn   = w & 1;       // 0..1
  const int m0   = blockIdx.y * BM;
  const int n0   = blockIdx.x * BN;

  // per-thread staging slots (16B async copies)
  const int rA = tid >> 1, cA = (tid & 1) * 16;    // A: 2x b128 per thread
  const int rW = tid >> 2, cW = (tid & 3) * 8;     // W: 1x b128 per thread
  unsigned aLds[2], wLds[2];
  aLds[0] = lds_off(&lsA[0][rA * LDT + cA]);
  aLds[1] = lds_off(&lsA[1][rA * LDT + cA]);
  wLds[0] = lds_off(&lsW[0][rW * LDT + cW]);
  wLds[1] = lds_off(&lsW[1][rW * LDT + cW]);
  unsigned aOff = (unsigned)(((size_t)(m0 + rA) * K + cA) * 2);
  unsigned wOff = (unsigned)(((size_t)(n0 + rW) * K + cW) * 2);

  v8f acc[2][2];
#pragma unroll
  for (int i = 0; i < 2; ++i)
#pragma unroll
    for (int j = 0; j < 2; ++j) acc[i][j] = (v8f){};

  // prologue: stage chunk 0 into buffer 0
  async_ld_b128(aLds[0],      aOff,       A);
  async_ld_b128(aLds[0] + 16, aOff + 16u, A);
  async_ld_b128(wLds[0],      wOff,       W);
  aOff += BK * 2; wOff += BK * 2;
  wait_async0();
  __syncthreads();

  int buf = 0;
  for (int k0 = 0; k0 < K; k0 += BK) {
    if (k0 + BK < K) {   // stage next chunk into the other buffer
      const int nb = buf ^ 1;
      async_ld_b128(aLds[nb],      aOff,       A);
      async_ld_b128(aLds[nb] + 16, aOff + 16u, A);
      async_ld_b128(wLds[nb],      wOff,       W);
      aOff += BK * 2; wOff += BK * 2;
    }

    v16bf af[2], wf[2];
#pragma unroll
    for (int i = 0; i < 2; ++i)
      af[i] = load_frag_A(&lsA[buf][(wm * 32 + i * 16) * LDT], LDT, lane);
#pragma unroll
    for (int j = 0; j < 2; ++j)
      wf[j] = load_frag_B(&lsW[buf][(wn * 32 + j * 16) * LDT], LDT, lane);

#pragma unroll
    for (int i = 0; i < 2; ++i)
#pragma unroll
      for (int j = 0; j < 2; ++j)
        acc[i][j] = __builtin_amdgcn_wmma_f32_16x16x32_bf16(
            false, af[i], false, wf[j], (short)0, acc[i][j], false, false);

    wait_async0();        // next buffer landed
    __syncthreads();      // everyone done reading current buffer
    buf ^= 1;
  }

  // epilogue: C/D layout -> lane holds (M = (lane>>4)*8 + v, N = lane&15)
#pragma unroll
  for (int i = 0; i < 2; ++i)
#pragma unroll
    for (int j = 0; j < 2; ++j) {
      const int ncol = n0 + wn * 32 + j * 16 + (lane & 15);
      const float bv = bias[ncol];
#pragma unroll
      for (int v = 0; v < 8; ++v) {
        const int mrow = m0 + wm * 32 + i * 16 + (lane >> 4) * 8 + v;
        C[(size_t)mrow * N + ncol] = acc[i][j][v] + bv;
      }
    }
}

// ---------------------------------------------------------------------------
// Persistent recurrence: h_t = tanh(xp_t + h_{t-1} @ W_hh^T)
// WG g owns hidden cols [g*64, g*64+64). Its W_hh slice (64x1024 bf16) is
// DMA'd into LDS by ONE tensor_load_to_lds (TDM) with LDS padding: 16B pad
// every 512 bf16 -> row stride 2080B (16B-aligned, conflict-mitigated).
// h double-buffered in global bf16; one grid barrier per step.
// ---------------------------------------------------------------------------
#define RROW 2080u   // LDS bytes per W row after TDM padding

__device__ __forceinline__ v16bf load_frag_B_rec(const __bf16* lsW, int nbase,
                                                 int k0, int lane) {
  const int half = lane >> 4;
  const int row  = nbase + (lane & 15);
  const unsigned off = (unsigned)row * RROW + (unsigned)(k0 + half * 16) * 2u +
                       ((k0 >= 512) ? 16u : 0u);
  const __bf16* p = (const __bf16*)((const char*)lsW + off);
  return make_v16(*(const v8bf*)(p), *(const v8bf*)(p + 8));
}

__global__ __launch_bounds__(256)
void rnn_recurrence(const float* __restrict__ xp, const __bf16* __restrict__ Whh,
                    __bf16* __restrict__ hseq, __bf16* __restrict__ hbuf,
                    unsigned* __restrict__ ctr) {
  extern __shared__ __align__(16) __bf16 lsW[];   // 64 rows x RROW bytes
  const int tid  = threadIdx.x;
  const int lane = tid & 31;
  const int w    = tid >> 5;
  const int mi   = w >> 2;          // 0..1 batch tile
  const int ni   = w & 3;           // 0..3 n tile within slice
  const int g    = blockIdx.x;      // slice id

  // --- TDM: one descriptor pulls the 64x1024 bf16 W_hh slice into LDS ---
  if (w == 0) {
    const unsigned long long ga =
        (unsigned long long)(uintptr_t)(Whh + (size_t)g * 64 * HID_);
    u32x4 g0;
    g0.x = 1u;                                   // count=1 (valid descriptor)
    g0.y = lds_off(lsW);                         // lds_addr
    g0.z = (unsigned)(ga & 0xFFFFFFFFu);         // global_addr[31:0]
    g0.w = (unsigned)((ga >> 32) & 0x1FFFFFFu)   // global_addr[56:32]
           | (2u << 30);                         // type=2 ("image")
    u32x8 g1;
    g1[0] = (1u << 16)        // data_size = 2 bytes
          | (1u << 20)        // pad_enable
          | (7u << 22)        // pad_interval: 256 DWORDs (=512 bf16)
          | (3u << 25);       // pad_amount: 4 DWORDs (=16B)
    g1[1] = (unsigned)(HID_ & 0xFFFF) << 16;     // tensor_dim0 lo16 (1024)
    g1[2] = ((unsigned)HID_ >> 16)               // tensor_dim0 hi16
          | (64u << 16);                         // tensor_dim1 lo16 (64 rows)
    g1[3] = 0u | ((unsigned)HID_ << 16);         // tensor_dim1 hi16 | tile_dim0=1024
    g1[4] = 64u;                                 // tile_dim1=64 | tile_dim2=0
    g1[5] = (unsigned)HID_;                      // tensor_dim0_stride lo32
    g1[6] = 0u;                                  // stride hi16 | dim1_stride lo16
    g1[7] = 0u;
    asm volatile("tensor_load_to_lds %0, %1" :: "s"(g0), "s"(g1) : "memory");
    wait_tensor0();
  }
  __syncthreads();

  for (int t = 0; t < S_; ++t) {
    const __bf16* hrd = hbuf + (size_t)((t & 1) ^ 1) * B_ * HID_; // h_{t-1}
    __bf16*       hwr = hbuf + (size_t)(t & 1) * B_ * HID_;       // h_t

    v8f acc = (v8f){};
    if (t > 0) {
      for (int k0 = 0; k0 < HID_; k0 += 32) {
        v16bf af = load_frag_A(hrd + (size_t)(mi * 16) * HID_ + k0, HID_, lane);
        v16bf wf = load_frag_B_rec(lsW, ni * 16, k0, lane);
        acc = __builtin_amdgcn_wmma_f32_16x16x32_bf16(
            false, af, false, wf, (short)0, acc, false, false);
      }
    }

    const int nl   = ni * 16 + (lane & 15);
    const int hcol = g * 64 + nl;
#pragma unroll
    for (int v = 0; v < 8; ++v) {
      const int b = mi * 16 + (lane >> 4) * 8 + v;
      float val = acc[v] + xp[((size_t)b * S_ + t) * HID_ + hcol];
      val = tanhf(val);
      const __bf16 hb = (__bf16)val;
      hwr[(size_t)b * HID_ + hcol]             = hb;
      hseq[((size_t)b * S_ + t) * HID_ + hcol] = hb;
    }
    if (t + 1 < S_) {  // hint next step's xp line toward the caches
      const int b0r = mi * 16 + (lane >> 4) * 8;
      __builtin_prefetch(&xp[((size_t)b0r * S_ + t + 1) * HID_ + hcol], 0, 1);
    }

    // grid barrier across the 16 WGs
    __syncthreads();
    if (tid == 0) {
      __hip_atomic_fetch_add(&ctr[t], 1u, __ATOMIC_ACQ_REL,
                             __HIP_MEMORY_SCOPE_AGENT);
      while (__hip_atomic_load(&ctr[t], __ATOMIC_ACQUIRE,
                               __HIP_MEMORY_SCOPE_AGENT) < (unsigned)NWG_REC) {
        __builtin_amdgcn_s_sleep(1);
      }
    }
    __syncthreads();
  }
}

// ---------------------------------------------------------------------------
// Small helper kernels
// ---------------------------------------------------------------------------
__global__ void f32_to_bf16_kernel(const float* __restrict__ src,
                                   __bf16* __restrict__ dst, int n) {
  const int i = blockIdx.x * 256 + threadIdx.x;
  if (i < n) dst[i] = (__bf16)src[i];
}

__global__ void make_bias_kernel(const float* bi0, const float* bh0,
                                 const float* bi1, const float* bh1,
                                 float* b0, float* b1) {
  const int i = blockIdx.x * 256 + threadIdx.x;
  if (i < HID_) { b0[i] = bi0[i] + bh0[i]; b1[i] = bi1[i] + bh1[i]; }
}

__global__ __launch_bounds__(256)
void gather_emb_kernel(const int* __restrict__ tokens,
                       const float* __restrict__ emb,
                       __bf16* __restrict__ xb) {
  const int row = blockIdx.x;                 // b*S + s
  const int tok = tokens[row];
  const float* src = emb + (size_t)tok * EMB_;
  __bf16*      dst = xb  + (size_t)row * EMB_;
  for (int e = threadIdx.x; e < EMB_; e += 256) dst[e] = (__bf16)src[e];
}

__global__ void zero_ctr_kernel(unsigned* c, int n) {
  const int i = blockIdx.x * 256 + threadIdx.x;
  if (i < n) c[i] = 0u;
}

// ---------------------------------------------------------------------------
extern "C" void kernel_launch(void* const* d_in, const int* in_sizes, int n_in,
                              void* d_out, int out_size, void* d_ws, size_t ws_size,
                              hipStream_t stream) {
  (void)in_sizes; (void)n_in; (void)out_size; (void)ws_size;
  const int*   tokens = (const int*)  d_in[0];
  const float* emb    = (const float*)d_in[1];
  const float* W_ih0  = (const float*)d_in[2];
  const float* W_hh0  = (const float*)d_in[3];
  const float* b_ih0  = (const float*)d_in[4];
  const float* b_hh0  = (const float*)d_in[5];
  const float* W_ih1  = (const float*)d_in[6];
  const float* W_hh1  = (const float*)d_in[7];
  const float* b_ih1  = (const float*)d_in[8];
  const float* b_hh1  = (const float*)d_in[9];
  const float* W_out  = (const float*)d_in[10];
  const float* b_out  = (const float*)d_in[11];
  float* out = (float*)d_out;

  // ---- workspace carve (256B aligned) ----
  char* base = (char*)d_ws;
  size_t off = 0;
  auto carve = [&](size_t bytes) {
    char* p = base + off;
    off = (off + bytes + 255) & ~(size_t)255;
    return p;
  };
  const size_t M = (size_t)B_ * S_;                       // 16384
  __bf16*  xb     = (__bf16*)carve(M * EMB_ * 2);         // 16 MB
  __bf16*  h0b    = (__bf16*)carve(M * HID_ * 2);         // 32 MB
  __bf16*  h1b    = (__bf16*)carve(M * HID_ * 2);         // 32 MB
  float*   xp     = (float*) carve(M * HID_ * 4);         // 64 MB
  __bf16*  wih0b  = (__bf16*)carve((size_t)HID_ * EMB_ * 2);
  __bf16*  whh0b  = (__bf16*)carve((size_t)HID_ * HID_ * 2);
  __bf16*  wih1b  = (__bf16*)carve((size_t)HID_ * HID_ * 2);
  __bf16*  whh1b  = (__bf16*)carve((size_t)HID_ * HID_ * 2);
  __bf16*  woutb  = (__bf16*)carve((size_t)LAB_ * HID_ * 2);
  float*   b0     = (float*) carve(HID_ * 4);
  float*   b1     = (float*) carve(HID_ * 4);
  __bf16*  hbuf   = (__bf16*)carve((size_t)2 * B_ * HID_ * 2); // double buffer
  unsigned* ctr   = (unsigned*)carve(S_ * 4);

  // ---- 1. weight conversion + bias combine ----
  auto cvt = [&](const float* s, __bf16* d, int n) {
    f32_to_bf16_kernel<<<(n + 255) / 256, 256, 0, stream>>>(s, d, n);
  };
  cvt(W_ih0, wih0b, HID_ * EMB_);
  cvt(W_hh0, whh0b, HID_ * HID_);
  cvt(W_ih1, wih1b, HID_ * HID_);
  cvt(W_hh1, whh1b, HID_ * HID_);
  cvt(W_out, woutb, LAB_ * HID_);
  make_bias_kernel<<<(HID_ + 255) / 256, 256, 0, stream>>>(b_ih0, b_hh0,
                                                           b_ih1, b_hh1, b0, b1);

  // ---- 2. embedding gather ----
  gather_emb_kernel<<<(int)M, 256, 0, stream>>>(tokens, emb, xb);

  const size_t recLds = (size_t)64 * RROW;   // 133120 B dynamic LDS

  // ---- 3. layer 0 input projection ----
  gemm_bf16_wmma<<<dim3(HID_ / BN, (int)(M / BM)), 256, 0, stream>>>(
      xb, wih0b, b0, xp, (int)M, HID_, EMB_);

  // ---- 4. layer 0 recurrence ----
  zero_ctr_kernel<<<(S_ + 255) / 256, 256, 0, stream>>>(ctr, S_);
  rnn_recurrence<<<NWG_REC, 256, recLds, stream>>>(xp, whh0b, h0b, hbuf, ctr);

  // ---- 5. layer 1 input projection ----
  gemm_bf16_wmma<<<dim3(HID_ / BN, (int)(M / BM)), 256, 0, stream>>>(
      h0b, wih1b, b1, xp, (int)M, HID_, HID_);

  // ---- 6. layer 1 recurrence ----
  zero_ctr_kernel<<<(S_ + 255) / 256, 256, 0, stream>>>(ctr, S_);
  rnn_recurrence<<<NWG_REC, 256, recLds, stream>>>(xp, whh1b, h1b, hbuf, ctr);

  // ---- 7. output head ----
  gemm_bf16_wmma<<<dim3(LAB_ / BN, (int)(M / BM)), 256, 0, stream>>>(
      h1b, woutb, b_out, out, (int)M, LAB_, HID_);
}
